// BRNNEncoder_60370060312702
// MI455X (gfx1250) — matrix operand order repaired and checked by reading, not compiled, not run on hardware
//
#include <hip/hip_runtime.h>
#include <hip/hip_bf16.h>
#include <math.h>

typedef __bf16 bf16_t;
typedef __attribute__((ext_vector_type(16))) __bf16 v16bf;
typedef __attribute__((ext_vector_type(8)))  float  v8f;
typedef __attribute__((ext_vector_type(4)))  unsigned int v4u;
typedef __attribute__((ext_vector_type(8)))  int  v8i;
typedef __attribute__((ext_vector_type(4)))  int  v4i;

namespace {
constexpr int kE = 512;
constexpr int kH = 512;
constexpr int kB = 32;
constexpr int kT = 512;
constexpr int kM = kT * kB;            // 16384 GEMM rows

// RNN scan: split U between LDS and registers (bf16 U is 512KB > 320KB LDS)
constexpr int kLdsK    = 224;          // K rows of U kept in LDS (7 chunks of 32)
constexpr int kRegC    = (kH - kLdsK) / 32;  // 9 K-chunks of U kept in VGPRs
constexpr int kUStride = kLdsK + 2;    // LDS row stride (226) -> conflict-free
constexpr int kUGStr   = kH + 2;       // ubfT global row stride (514)
constexpr int kHStride = kH + 2;

union BF16x16 { v16bf v; unsigned u[8]; __bf16 h[16]; };
union F32x8   { v8f  v; float f[8]; };

__device__ inline v8f wmma_bf16(v16bf a, v16bf b, v8f c) {
  // (neg_a, A, neg_b, B, c_mod, C, reuse_a, reuse_b)
  return __builtin_amdgcn_wmma_f32_16x16x32_bf16(false, a, false, b,
                                                 (short)0, c, false, false);
}

#if __has_builtin(__builtin_amdgcn_tensor_load_to_lds)
#define HAVE_TDM 1
// 2D tile load via Tensor Data Mover (amdgpu-toolchain 6-arg builtin).
// D# packing per CDNA5 ISA ch.8:
//  group0: count=1 | lds_addr | global_addr | type=2
//  group1: data_size=2B, optional LDS pad (16 dwords -> +1 dword),
//          tensor_dim0/1, tile_dim0/1, tensor_dim0_stride (elements)
//  groups 2/3 + trailing group: unused for 2D tiles (zero-filled)
__device__ inline void tdm_load_2d(unsigned lds_off, unsigned long long gaddr,
                                   unsigned td0, unsigned td1,
                                   unsigned tile0, unsigned tile1,
                                   unsigned stride0, bool pad16) {
  v4u g0; v8i g1;
  v4i z4; z4[0] = 0; z4[1] = 0; z4[2] = 0; z4[3] = 0;
  v8i z8; z8[0] = 0; z8[1] = 0; z8[2] = 0; z8[3] = 0;
          z8[4] = 0; z8[5] = 0; z8[6] = 0; z8[7] = 0;
  g0[0] = 1u;                                   // count=1, user mode
  g0[1] = lds_off;                              // lds_addr (bytes)
  g0[2] = (unsigned)gaddr;                      // global_addr[31:0]
  g0[3] = (unsigned)(gaddr >> 32) | (2u << 30); // global_addr[56:32] | type=2
  g1[0] = (int)((1u << 16) |                    // data_size = 2 bytes
                (pad16 ? ((1u << 20) | (3u << 22)) : 0u)); // pad en,int=16dw,amt=1dw
  g1[1] = (int)((td0 & 0xFFFFu) << 16);                    // tensor_dim0 lo
  g1[2] = (int)((td0 >> 16) | ((td1 & 0xFFFFu) << 16));    // td0 hi | td1 lo
  g1[3] = (int)((td1 >> 16) | (tile0 << 16));              // td1 hi | tile_dim0
  g1[4] = (int)tile1;                                      // tile_dim1 (tile2=0)
  g1[5] = (int)stride0;                                    // tensor_dim0_stride lo
  g1[6] = 0; g1[7] = 0;
  __builtin_amdgcn_tensor_load_to_lds(g0, g1, z4, z4, z8, 0);
}
#else
#define HAVE_TDM 0
#endif
} // namespace

// ---------------------------------------------------------------------------
// Prep 1: gather + convert embeddings: ebf[m][k] = bf16(emb[x[b][t]][k]),
// m = t*B+b.  8.4M elems, pure bandwidth.
// ---------------------------------------------------------------------------
__global__ __launch_bounds__(256) void brnn_gather_cvt(
    const int* __restrict__ x, const float* __restrict__ emb,
    bf16_t* __restrict__ ebf)
{
  const size_t idx0 = ((size_t)blockIdx.x * 256 + threadIdx.x) * 8;
  const int m = (int)(idx0 >> 9);
  const int k = (int)(idx0 & 511);
  const int tokid = x[(m & 31) * kT + (m >> 5)];      // x is (B,T) row-major
  const float* __restrict__ src = emb + (size_t)tokid * kE + k;
#pragma unroll
  for (int i = 0; i < 8; ++i) ebf[idx0 + i] = (bf16_t)src[i];
}

// ---------------------------------------------------------------------------
// Prep 2: transpose + convert a 512x512 f32 matrix: out[c][K] = bf16(in[K][c])
// (used for W -> wbfT stride 512, and U -> ubfT stride 514).
// ---------------------------------------------------------------------------
__global__ __launch_bounds__(256) void brnn_transpose_cvt(
    const float* __restrict__ in, bf16_t* __restrict__ outp, int ostride)
{
  const int idx0 = (blockIdx.x * 256 + threadIdx.x) * 8;
  const int c  = idx0 >> 9;
  const int K0 = idx0 & 511;
#pragma unroll
  for (int i = 0; i < 8; ++i)
    outp[(size_t)c * ostride + K0 + i] = (bf16_t)in[(size_t)(K0 + i) * kH + c];
}

// ---------------------------------------------------------------------------
// Projection GEMM: xproj[m][h] = ebf[m] @ W + bias (per direction).
// grid (M/128, H/64, 2), block 256 = 8 waves; each wave: 16 rows x 64 cols.
// A/B tiles are DMA'd into double-buffered padded LDS by the TDM (wave 0),
// overlapped with WMMA compute; s_wait_tensorcnt gates each chunk.
// ---------------------------------------------------------------------------
__global__ __launch_bounds__(256) void brnn_proj_wmma(
    const bf16_t* __restrict__ ebf,
    const bf16_t* __restrict__ wbfTf, const bf16_t* __restrict__ wbfTb,
    const float* __restrict__ bias_f, const float* __restrict__ bias_b,
    float* __restrict__ xf, float* __restrict__ xb)
{
  const int dir = blockIdx.z;
  const bf16_t* __restrict__ Wt   = dir ? wbfTb : wbfTf;
  const float*  __restrict__ bias = dir ? bias_b : bias_f;
  float* __restrict__ xout        = dir ? xb : xf;

  __shared__ bf16_t As[2][128][34];  // [buf][row][k], +1dw pad per row
  __shared__ bf16_t Bs[2][64][34];   // [buf][col][k], +1dw pad per row

  const int tid   = threadIdx.x;
  const int wave  = tid >> 5;
  const int lane  = tid & 31;
  const int mBase = blockIdx.x * 128;
  const int nBase = blockIdx.y * 64;

  F32x8 acc[4];
#pragma unroll
  for (int i = 0; i < 4; ++i)
#pragma unroll
    for (int j = 0; j < 8; ++j) acc[i].f[j] = 0.f;

  const int hrow = lane & 15;
  const int kbA  = (lane < 16) ? 0 : 16;   // A-matrix K base (contiguous, matches B)
  const int kbB  = (lane < 16) ? 0 : 16;   // B-matrix K base

  const unsigned long long aBase =
      (unsigned long long)(uintptr_t)(ebf + (size_t)mBase * kE);
  const unsigned long long bBase =
      (unsigned long long)(uintptr_t)(Wt + (size_t)nBase * kH);

#if HAVE_TDM
  if (wave == 0) {   // preload chunk 0
    tdm_load_2d((unsigned)(uintptr_t)&As[0][0][0], aBase, kE, kM, 32, 128, kE, true);
    tdm_load_2d((unsigned)(uintptr_t)&Bs[0][0][0], bBase, kH, kH, 32, 64,  kH, true);
  }
#endif

  for (int kc = 0; kc < kE / 32; ++kc) {
    const int buf = kc & 1;
#if HAVE_TDM
    if (wave == 0) {
      if (kc + 1 < kE / 32) {   // prefetch next chunk into the other buffer
        const unsigned k1 = (unsigned)(kc + 1) * 32;
        tdm_load_2d((unsigned)(uintptr_t)&As[buf ^ 1][0][0],
                    aBase + (unsigned long long)k1 * 2, kE, kM, 32, 128, kE, true);
        tdm_load_2d((unsigned)(uintptr_t)&Bs[buf ^ 1][0][0],
                    bBase + (unsigned long long)k1 * 2, kH, kH, 32, 64, kH, true);
        __builtin_amdgcn_s_wait_tensorcnt(2);   // chunk kc complete
      } else {
        __builtin_amdgcn_s_wait_tensorcnt(0);
      }
    }
#else
    {   // fallback: cooperative thread copy (no TDM builtin on this toolchain)
      const int k0 = kc * 32;
#pragma unroll
      for (int i = 0; i < 16; ++i) {
        const int idx = i * 256 + tid, r = idx >> 5, k = idx & 31;
        As[buf][r][k] = ebf[(size_t)(mBase + r) * kE + k0 + k];
      }
#pragma unroll
      for (int i = 0; i < 8; ++i) {
        const int idx = i * 256 + tid, n = idx & 63, k = idx >> 6;
        Bs[buf][n][k] = Wt[(size_t)(nBase + n) * kH + k0 + k];
      }
    }
#endif
    __syncthreads();

    // A fragment (16x32): lanes 0-15 K{0..15}, lanes 16-31 K{16..31}
    BF16x16 a;
    const int r = wave * 16 + hrow;
#pragma unroll
    for (int p = 0; p < 8; ++p) {
      const int kk = kbA + 2 * p;
      a.u[p] = *(const unsigned*)&As[buf][r][kk];
    }
#pragma unroll
    for (int ns = 0; ns < 4; ++ns) {
      BF16x16 bm;
      const int n = ns * 16 + hrow;
#pragma unroll
      for (int p = 0; p < 8; ++p)
        bm.u[p] = *(const unsigned*)&Bs[buf][n][kbB + 2 * p];
      acc[ns].v = wmma_bf16(a.v, bm.v, acc[ns].v);
    }
    __syncthreads();
  }

  const int moff = (lane < 16) ? 0 : 8;   // C/D layout: lane>=16 holds M=j+8
#pragma unroll
  for (int ns = 0; ns < 4; ++ns) {
    const int col = nBase + ns * 16 + hrow;
    const float bv = bias[col];
#pragma unroll
    for (int j = 0; j < 8; ++j) {
      const int m = mBase + wave * 16 + moff + j;
      xout[(size_t)m * kH + col] = acc[ns].f[j] + bv;
    }
  }
}

// ---------------------------------------------------------------------------
// Recurrence: persistent workgroup per direction (grid = 2, block = 1024).
// U panel (K<224) DMA'd once into LDS by TDM; K>=224 kept as register
// B-fragments; h double-buffered bf16 in LDS; one s_barrier per step.
// ---------------------------------------------------------------------------
__global__ __launch_bounds__(1024, 1) void brnn_scan_wmma(
    const float* __restrict__ xf, const float* __restrict__ xb,
    const float* __restrict__ h0f, const float* __restrict__ h0b,
    const bf16_t* __restrict__ ubfTf, const bf16_t* __restrict__ ubfTb,
    float* __restrict__ out)
{
  extern __shared__ char smem[];
  bf16_t* __restrict__ Us = (bf16_t*)smem;                         // [kH][kUStride]
  bf16_t* __restrict__ hb =
      (bf16_t*)(smem + (size_t)kH * kUStride * sizeof(bf16_t));    // [2][kB][kHStride]

  const int dir = blockIdx.x;
  const float*  __restrict__ xp   = dir ? xb : xf;
  const bf16_t* __restrict__ ubfT = dir ? ubfTb : ubfTf;
  const float*  __restrict__ h0   = dir ? h0b : h0f;
  float* __restrict__ stateOut =
      out + (size_t)kB * kT * 2 * kH + (size_t)dir * kB * kH;

  const int tid  = threadIdx.x;
  const int wave = tid >> 5;
  const int lane = tid & 31;
  const int hrow = lane & 15;
  const int col  = wave * 16 + hrow;       // this wave's output column
  const int kbB  = (lane < 16) ? 0 : 16;
  const int kbA  = (lane < 16) ? 0 : 16;
  const int moff = (lane < 16) ? 0 : 8;

  // one-time: U[K<224] panel -> LDS.  ubfT is [col][K] bf16 with row stride
  // 514, LDS rows are 226 elements: a single 2D TDM descriptor.
#if HAVE_TDM
  if (wave == 0) {
    tdm_load_2d((unsigned)(uintptr_t)Us, (unsigned long long)(uintptr_t)ubfT,
                kUGStr, kH, kUStride, kH, kUGStr, false);
    __builtin_amdgcn_s_wait_tensorcnt(0);
  }
#else
  for (int idx = tid; idx < kH * kUStride; idx += 1024) {
    const int c = idx / kUStride, K = idx % kUStride;
    Us[idx] = ubfT[(size_t)c * kUGStr + K];
  }
#endif

  // one-time: U[K>=224] -> registers, already in WMMA B-fragment layout.
  BF16x16 ufrag[kRegC];
  {
    const bf16_t* __restrict__ ucol = ubfT + (size_t)col * kUGStr;
#pragma unroll
    for (int kc = 0; kc < kRegC; ++kc)
#pragma unroll
      for (int p = 0; p < 8; ++p)
        ufrag[kc].u[p] = *(const unsigned*)&ucol[kLdsK + kc * 32 + kbB + 2 * p];
  }
  // initial hidden state -> LDS buffer 0 (bf16)
  for (int idx = tid; idx < kB * kH; idx += 1024) {
    const int b = idx >> 9, c = idx & (kH - 1);
    hb[(size_t)b * kHStride + c] = (bf16_t)h0[idx];
  }
  __syncthreads();

  for (int s = 0; s < kT; ++s) {
    const int t = dir ? (kT - 1 - s) : s;
    const bf16_t* __restrict__ hcur = hb + (size_t)(s & 1) * kB * kHStride;
    bf16_t* __restrict__ hnxt = hb + (size_t)((s & 1) ^ 1) * kB * kHStride;

    F32x8 acc0, acc1;
#pragma unroll
    for (int j = 0; j < 8; ++j) { acc0.f[j] = 0.f; acc1.f[j] = 0.f; }

#pragma unroll
    for (int kc = 0; kc < kH / 32; ++kc) {
      BF16x16 a0, a1, bm;
#pragma unroll
      for (int p = 0; p < 8; ++p) {
        const int kk = kc * 32 + kbA + 2 * p;
        a0.u[p] = *(const unsigned*)&hcur[(size_t)hrow * kHStride + kk];
        a1.u[p] = *(const unsigned*)&hcur[(size_t)(16 + hrow) * kHStride + kk];
      }
      if (kc < kLdsK / 32) {
#pragma unroll
        for (int p = 0; p < 8; ++p)
          bm.u[p] =
              *(const unsigned*)&Us[(size_t)col * kUStride + kc * 32 + kbB + 2 * p];
      } else {
        bm = ufrag[kc - kLdsK / 32];
      }
      acc0.v = wmma_bf16(a0.v, bm.v, acc0.v);
      acc1.v = wmma_bf16(a1.v, bm.v, acc1.v);
    }

    const float* __restrict__ xrow = xp + (size_t)t * kB * kH;
    if (s + 1 < kT) {   // warm L2/WGP$ for next step's 64KB xproj slab
      const int tn = dir ? (t - 1) : (t + 1);
      __builtin_prefetch(xp + (size_t)tn * kB * kH + tid * 16, 0, 0);
    }
#pragma unroll
    for (int j = 0; j < 8; ++j) {
      {
        const int b = moff + j;                      // M-tile 0: rows 0..15
        const float v = tanhf(acc0.f[j] + xrow[(size_t)b * kH + col]);
        hnxt[(size_t)b * kHStride + col] = (bf16_t)v;
        out[((size_t)b * kT + t) * (2 * kH) + (size_t)dir * kH + col] = v;
        if (s == kT - 1) stateOut[b * kH + col] = v;
      }
      {
        const int b = 16 + moff + j;                 // M-tile 1: rows 16..31
        const float v = tanhf(acc1.f[j] + xrow[(size_t)b * kH + col]);
        hnxt[(size_t)b * kHStride + col] = (bf16_t)v;
        out[((size_t)b * kT + t) * (2 * kH) + (size_t)dir * kH + col] = v;
        if (s == kT - 1) stateOut[b * kH + col] = v;
      }
    }
    __syncthreads();
  }
}

// ---------------------------------------------------------------------------
extern "C" void kernel_launch(void* const* d_in, const int* in_sizes, int n_in,
                              void* d_out, int out_size, void* d_ws, size_t ws_size,
                              hipStream_t stream) {
  const int*   x   = (const int*)  d_in[0];
  const float* h0f = (const float*)d_in[1];
  const float* h0b = (const float*)d_in[2];
  const float* emb = (const float*)d_in[3];
  const float* Wf  = (const float*)d_in[4];
  const float* Uf  = (const float*)d_in[5];
  const float* bfv = (const float*)d_in[6];
  const float* Wb  = (const float*)d_in[7];
  const float* Ub  = (const float*)d_in[8];
  const float* bbv = (const float*)d_in[9];
  float* out = (float*)d_out;

  // workspace layout
  float*  xf    = (float*)d_ws;                               // 32 MB
  float*  xb    = xf + (size_t)kM * kH;                       // 32 MB
  bf16_t* ebf   = (bf16_t*)(xb + (size_t)kM * kH);            // 16 MB
  bf16_t* wbfTf = ebf + (size_t)kM * kE;                      // 512 KB
  bf16_t* wbfTb = wbfTf + (size_t)kH * kH;                    // 512 KB
  bf16_t* ubfTf = wbfTb + (size_t)kH * kH;                    // 514 KB
  bf16_t* ubfTb = ubfTf + (size_t)kH * kUGStr;                // 514 KB

  // prep: gather/convert embeddings; transpose/convert W and U to bf16
  brnn_gather_cvt<<<(kM * kE) / (256 * 8), 256, 0, stream>>>(x, emb, ebf);
  const int tgrid = (kH * kH) / (256 * 8);
  brnn_transpose_cvt<<<tgrid, 256, 0, stream>>>(Wf, wbfTf, kH);
  brnn_transpose_cvt<<<tgrid, 256, 0, stream>>>(Wb, wbfTb, kH);
  brnn_transpose_cvt<<<tgrid, 256, 0, stream>>>(Uf, ubfTf, kUGStr);
  brnn_transpose_cvt<<<tgrid, 256, 0, stream>>>(Ub, ubfTb, kUGStr);

  dim3 pgrid(kM / 128, kH / 64, 2);
  brnn_proj_wmma<<<pgrid, 256, 0, stream>>>(ebf, wbfTf, wbfTb,
                                            bfv, bbv, xf, xb);

  const size_t scanLds = (size_t)kH * kUStride * sizeof(bf16_t) +
                         (size_t)2 * kB * kHStride * sizeof(bf16_t);  // ~297 KB
  (void)hipFuncSetAttribute((const void*)brnn_scan_wmma,
                            hipFuncAttributeMaxDynamicSharedMemorySize,
                            (int)scanLds);
  brnn_scan_wmma<<<dim3(2), 1024, scanLds, stream>>>(xf, xb, h0f, h0b,
                                                     ubfTf, ubfTb, out);
}
